// TriangularSelfAttention_54211077210611
// MI455X (gfx1250) — matrix operand order repaired and checked by reading, not compiled, not run on hardware
//
#include <hip/hip_runtime.h>

#define E_DIM 128
#define CH    128   // C*H
#define CDIM  32
#define NH    4
#define LSEQ  256
#define QSCALE 0.17677669529663687f  // 1/sqrt(32)

typedef __attribute__((ext_vector_type(16))) _Float16 v16h;
typedef __attribute__((ext_vector_type(8)))  _Float16 v8h;
typedef __attribute__((ext_vector_type(8)))  float    v8f;

__device__ __forceinline__ float wave_sum32(float v){
#pragma unroll
  for (int off = 16; off > 0; off >>= 1) v += __shfl_xor(v, off, 32);
  return v;
}
__device__ __forceinline__ float wave_max32(float v){
#pragma unroll
  for (int off = 16; off > 0; off >>= 1) v = fmaxf(v, __shfl_xor(v, off, 32));
  return v;
}

__device__ __forceinline__ v16h combine8(v8h lo, v8h hi){
  return __builtin_shufflevector(lo, hi, 0,1,2,3,4,5,6,7,8,9,10,11,12,13,14,15);
}

// A fragment (16x32 f16): lane holds row lane&15; elems 0..15 = K kb..kb+15,
// kb = (lane>>4)*16   (contiguous order, matches load_b_frag_T)
// Requires p 16B-aligned at every call site.
__device__ __forceinline__ v16h load_a_frag(const _Float16* base, int stride, int kofs, int lane){
  const _Float16* p = base + (size_t)(lane & 15) * stride + kofs + ((lane >> 4) << 4);
  v8h lo = *(const v8h*)p;
  v8h hi = *(const v8h*)(p + 8);
  return combine8(lo, hi);
}

// B fragment (32x16) where B[k][n] = src[(nbase+n)*stride + kbase + k]  (B = src^T)
// lane holds col n = lane&15, K = (lane>>4)*16 + i  -> one contiguous 32B read.
__device__ __forceinline__ v16h load_b_frag_T(const _Float16* src, int stride, int nbase, int kbase, int lane){
  const _Float16* p = src + (size_t)(nbase + (lane & 15)) * stride + kbase + ((lane >> 4) << 4);
  return *(const v16h*)p;
}

// ---------------- Kernel A: layernorm + 4 projections + pair bias ----------------
__global__ __launch_bounds__(256) void ln_proj_kernel(
    const float* __restrict__ pair, const float* __restrict__ gamma, const float* __restrict__ beta,
    const float* __restrict__ Wq, const float* __restrict__ Wk, const float* __restrict__ Wv,
    const float* __restrict__ Wfu, const float* __restrict__ Wb,
    _Float16* __restrict__ qo, _Float16* __restrict__ ko, _Float16* __restrict__ vo,
    _Float16* __restrict__ fo, float* __restrict__ biaso)
{
  extern __shared__ char smem[];
  _Float16* xs  = (_Float16*)smem;        // [128][128] f16 row-major (32 KB)
  _Float16* wst = xs  + 128 * 128;        // W^T: [n=128][k=128] f16   (32 KB)
  _Float16* osh = wst + 128 * 128;        // [128][128] f16 staging    (32 KB)

  const int tid = threadIdx.x, lane = tid & 31, wv = tid >> 5;
  const size_t rowbase = (size_t)blockIdx.x * 128;

  // ---- layernorm: each wave handles 16 rows, 4 cols per lane ----
  const int c0 = lane * 4;
  float g4[4], b4[4];
#pragma unroll
  for (int i = 0; i < 4; ++i){ g4[i] = gamma[c0 + i]; b4[i] = beta[c0 + i]; }
  for (int r = 0; r < 16; ++r){
    const int row = wv * 16 + r;
    const float* src = pair + (rowbase + row) * E_DIM + c0;
    __builtin_prefetch(src + E_DIM, 0, 0);
    float x0 = src[0], x1 = src[1], x2 = src[2], x3 = src[3];
    float mu = wave_sum32(x0 + x1 + x2 + x3) * (1.0f / 128.0f);
    float d0 = x0 - mu, d1 = x1 - mu, d2 = x2 - mu, d3 = x3 - mu;
    float var = wave_sum32(d0*d0 + d1*d1 + d2*d2 + d3*d3) * (1.0f / 128.0f);
    float rs = rsqrtf(var + 1e-5f);
    _Float16* dst = xs + row * 128 + c0;
    dst[0] = (_Float16)(d0 * rs * g4[0] + b4[0]);
    dst[1] = (_Float16)(d1 * rs * g4[1] + b4[1]);
    dst[2] = (_Float16)(d2 * rs * g4[2] + b4[2]);
    dst[3] = (_Float16)(d3 * rs * g4[3] + b4[3]);
  }
  __syncthreads();

  // ---- pair bias: b = x @ Wb  (128 rows x 4 heads, VALU dot) ----
#pragma unroll
  for (int m = 0; m < 2; ++m){
    int idx = tid * 2 + m;
    int row = idx >> 2, h = idx & 3;
    float acc = 0.f;
    for (int kk = 0; kk < 128; ++kk)
      acc += (float)xs[row * 128 + kk] * Wb[kk * 4 + h];
    biaso[(rowbase + row) * 4 + h] = acc;
  }

  // ---- 4 projections via WMMA f16 (f32 accum) ----
  const float scales[4] = {QSCALE, 1.f, 1.f, 1.f};
  const float* Ws[4] = {Wq, Wk, Wv, Wfu};
  _Float16* Os[4] = {qo, ko, vo, fo};

#pragma unroll
  for (int mtx = 0; mtx < 4; ++mtx){
    const float* W = Ws[mtx];
    // stage W transposed: wst[n][k] = W[k][n]  (f32 -> f16)
    for (int i = tid; i < 128 * 128 / 4; i += 256){
      float4 w4 = ((const float4*)W)[i];
      int kk = i >> 5, n0 = (i & 31) * 4;
      wst[(n0 + 0) * 128 + kk] = (_Float16)w4.x;
      wst[(n0 + 1) * 128 + kk] = (_Float16)w4.y;
      wst[(n0 + 2) * 128 + kk] = (_Float16)w4.z;
      wst[(n0 + 3) * 128 + kk] = (_Float16)w4.w;
    }
    __syncthreads();

    const _Float16* arow = xs + wv * 16 * 128;
    v16h afr[4];
#pragma unroll
    for (int kt = 0; kt < 4; ++kt) afr[kt] = load_a_frag(arow, 128, kt * 32, lane);

    const float sc = scales[mtx];
#pragma unroll
    for (int nt = 0; nt < 8; ++nt){
      v8f acc = {};
#pragma unroll
      for (int kt = 0; kt < 4; ++kt){
        v16h bfr = load_b_frag_T(wst, 128, nt * 16, kt * 32, lane);
        acc = __builtin_amdgcn_wmma_f32_16x16x32_f16(false, afr[kt], false, bfr,
                                                     (short)0, acc, false, false);
      }
#pragma unroll
      for (int j = 0; j < 8; ++j){
        int rr = wv * 16 + j + ((lane >> 4) << 3);
        osh[rr * 128 + nt * 16 + (lane & 15)] = (_Float16)(acc[j] * sc);
      }
    }
    __syncthreads();
    _Float16* dst = Os[mtx] + rowbase * 128;
    for (int i = tid; i < 128 * 128 / 8; i += 256)
      ((int4*)dst)[i] = ((const int4*)osh)[i];
    __syncthreads();
  }
}

// ---------------- Kernel B: attention per (s1, head, q-half) ----------------
__global__ __launch_bounds__(256) void attn_kernel(
    const _Float16* __restrict__ qf, const _Float16* __restrict__ kf, const _Float16* __restrict__ vf,
    const float* __restrict__ mask, const float* __restrict__ bias,
    _Float16* __restrict__ ao)
{
  extern __shared__ char smem[];
  _Float16* Ksh = (_Float16*)smem;            // [256][32] f16 (16 KB)
  _Float16* Qsh = Ksh + 256 * 32;             // [256][32]
  _Float16* Vt  = Qsh + 256 * 32;             // V^T: [c=32][v=256]
  float*    Ssc = (float*)(Vt + 32 * 256);    // [8 waves][16][256] f32 (128 KB)

  const int tid = threadIdx.x, lane = tid & 31, wv = tid >> 5;
  const int s1 = blockIdx.x, h = blockIdx.y, qh = blockIdx.z;

  // ---- stage K/Q via CDNA5 async loads (ASYNCcnt), V transposed via DS ----
  {
    const size_t base = (size_t)s1 * 256 * CH + h * CDIM + (size_t)tid * CH;
    unsigned ldsK = (unsigned)(size_t)(Ksh + tid * 32);
    unsigned ldsQ = (unsigned)(size_t)(Qsh + tid * 32);
    unsigned long long gk = (unsigned long long)(size_t)(kf + base);
    unsigned long long gq = (unsigned long long)(size_t)(qf + base);
    asm volatile(
      "global_load_async_to_lds_b128 %0, %1, off\n\t"
      "global_load_async_to_lds_b128 %0, %1, off offset:16\n\t"
      "global_load_async_to_lds_b128 %0, %1, off offset:32\n\t"
      "global_load_async_to_lds_b128 %0, %1, off offset:48\n\t"
      "global_load_async_to_lds_b128 %2, %3, off\n\t"
      "global_load_async_to_lds_b128 %2, %3, off offset:16\n\t"
      "global_load_async_to_lds_b128 %2, %3, off offset:32\n\t"
      "global_load_async_to_lds_b128 %2, %3, off offset:48"
      :: "v"(ldsK), "v"(gk), "v"(ldsQ), "v"(gq) : "memory");

    const v8h* vr = (const v8h*)(vf + base);
#pragma unroll
    for (int i = 0; i < 4; ++i){
      v8h x = vr[i];
#pragma unroll
      for (int u = 0; u < 8; ++u) Vt[(i * 8 + u) * 256 + tid] = x[u];
    }
    asm volatile("s_wait_asynccnt 0" ::: "memory");
  }
  __syncthreads();

  const int qoff = qh * 128 + wv * 16;
  float* Ssw = Ssc + wv * 16 * 256;

  // scores[q][v] = dot_c(k[s1,q,h,:], q[s1,v,h,:]) + mask[q,v] + bias[q,v,h]
  v16h aK = load_a_frag(Ksh + qoff * 32, 32, 0, lane);   // K=32: one WMMA per tile
#pragma unroll
  for (int vt = 0; vt < 16; ++vt){
    v16h bQ = load_b_frag_T(Qsh, 32, vt * 16, 0, lane);
    v8f acc = {};
    acc = __builtin_amdgcn_wmma_f32_16x16x32_f16(false, aK, false, bQ, (short)0, acc, false, false);
    const int vc = vt * 16 + (lane & 15);
#pragma unroll
    for (int j = 0; j < 8; ++j){
      int rloc = j + ((lane >> 4) << 3);
      int qr = qoff + rloc;
      float s = acc[j] + mask[qr * 256 + vc] + bias[((size_t)qr * 256 + vc) * 4 + h];
      Ssw[rloc * 256 + vc] = s;
    }
  }
  asm volatile("s_wait_dscnt 0" ::: "memory");   // wave-private scores ready

  // softmax over v: rowmax, then unnormalized exp -> f16 in place; keep 1/sum
  float rmax[16], rinv[16];
#pragma unroll
  for (int r = 0; r < 16; ++r){
    const float* row = Ssw + r * 256 + lane * 8;
    float m = -3.4e38f;
#pragma unroll
    for (int i = 0; i < 8; ++i) m = fmaxf(m, row[i]);
    rmax[r] = wave_max32(m);
  }
  _Float16* attnw = (_Float16*)Ssw;   // in-place; increasing-row order keeps aliasing safe
#pragma unroll
  for (int r = 0; r < 16; ++r){
    const float* row = Ssw + r * 256 + lane * 8;
    float e[8], s = 0.f;
#pragma unroll
    for (int i = 0; i < 8; ++i){ e[i] = exp2f((row[i] - rmax[r]) * 1.44269504f); s += e[i]; }
    rinv[r] = 1.0f / wave_sum32(s);
    asm volatile("s_wait_dscnt 0" ::: "memory");  // f32 row consumed before aliased f16 writes
    _Float16* drow = attnw + r * 256 + lane * 8;
#pragma unroll
    for (int i = 0; i < 8; ++i) drow[i] = (_Float16)e[i];
  }

  // out = attn @ V, scaled by 1/sum
#pragma unroll
  for (int nt = 0; nt < 2; ++nt){
    v8f acc = {};
#pragma unroll
    for (int kt = 0; kt < 8; ++kt){
      v16h aA = load_a_frag(attnw, 256, kt * 32, lane);
      v16h bV = load_b_frag_T(Vt, 256, nt * 16, kt * 32, lane);
      acc = __builtin_amdgcn_wmma_f32_16x16x32_f16(false, aA, false, bV, (short)0, acc, false, false);
    }
    const int cc = h * 32 + nt * 16 + (lane & 15);
#pragma unroll
    for (int j = 0; j < 8; ++j){
      int qr = qoff + j + ((lane >> 4) << 3);
      float val = acc[j] * ((lane & 16) ? rinv[j + 8] : rinv[j]);
      ao[((size_t)s1 * 256 + qr) * CH + cc] = (_Float16)val;
    }
  }
}

// ---------------- Kernel C: sigmoid gate + output projection ----------------
__global__ __launch_bounds__(256) void gate_out_kernel(
    const _Float16* __restrict__ ao, const _Float16* __restrict__ fo,
    const float* __restrict__ bfu, const float* __restrict__ Wo, const float* __restrict__ bout,
    float* __restrict__ out)
{
  extern __shared__ char smem[];
  _Float16* gsh = (_Float16*)smem;          // [128][128] f16
  _Float16* wst = gsh + 128 * 128;          // Wo^T: [n][k] f16
  float*    osh = (float*)(wst + 128 * 128);// [128][128] f32

  const int tid = threadIdx.x, lane = tid & 31, wv = tid >> 5;
  const size_t rowbase = (size_t)blockIdx.x * 128;

  // stage Wo transposed
  for (int i = tid; i < 128 * 128 / 4; i += 256){
    float4 w4 = ((const float4*)Wo)[i];
    int kk = i >> 5, n0 = (i & 31) * 4;
    wst[(n0 + 0) * 128 + kk] = (_Float16)w4.x;
    wst[(n0 + 1) * 128 + kk] = (_Float16)w4.y;
    wst[(n0 + 2) * 128 + kk] = (_Float16)w4.z;
    wst[(n0 + 3) * 128 + kk] = (_Float16)w4.w;
  }
  // gated input
  for (int i = tid; i < 128 * 128 / 4; i += 256){
    int col = (i * 4) & 127;
    const _Float16* a4 = ao + rowbase * 128 + i * 4;
    const _Float16* f4 = fo + rowbase * 128 + i * 4;
    _Float16* d = gsh + i * 4;
#pragma unroll
    for (int u = 0; u < 4; ++u){
      float fuv = (float)f4[u] + bfu[col + u];
      float gate = 1.0f / (1.0f + expf(-fuv));
      d[u] = (_Float16)((float)a4[u] * gate);
    }
  }
  __syncthreads();

  const _Float16* arow = gsh + wv * 16 * 128;
  v16h afr[4];
#pragma unroll
  for (int kt = 0; kt < 4; ++kt) afr[kt] = load_a_frag(arow, 128, kt * 32, lane);
#pragma unroll
  for (int nt = 0; nt < 8; ++nt){
    v8f acc = {};
#pragma unroll
    for (int kt = 0; kt < 4; ++kt){
      v16h bfr = load_b_frag_T(wst, 128, nt * 16, kt * 32, lane);
      acc = __builtin_amdgcn_wmma_f32_16x16x32_f16(false, afr[kt], false, bfr,
                                                   (short)0, acc, false, false);
    }
#pragma unroll
    for (int j = 0; j < 8; ++j){
      int rr = wv * 16 + j + ((lane >> 4) << 3);
      osh[rr * 128 + nt * 16 + (lane & 15)] = acc[j];
    }
  }
  __syncthreads();
  float* dst = out + rowbase * 128;
  for (int i = tid; i < 128 * 128 / 4; i += 256){
    float4 o4 = ((const float4*)osh)[i];
    int col = (i * 4) & 127;
    o4.x += bout[col]; o4.y += bout[col + 1]; o4.z += bout[col + 2]; o4.w += bout[col + 3];
    ((float4*)dst)[i] = o4;
  }
}

extern "C" void kernel_launch(void* const* d_in, const int* in_sizes, int n_in,
                              void* d_out, int out_size, void* d_ws, size_t ws_size,
                              hipStream_t stream)
{
  (void)in_sizes; (void)n_in; (void)out_size; (void)ws_size;
  const float* pair  = (const float*)d_in[0];
  const float* mask  = (const float*)d_in[1];
  const float* gamma = (const float*)d_in[2];
  const float* beta  = (const float*)d_in[3];
  const float* Wq    = (const float*)d_in[4];
  const float* Wk    = (const float*)d_in[5];
  const float* Wv    = (const float*)d_in[6];
  const float* Wb    = (const float*)d_in[7];
  const float* Wfu   = (const float*)d_in[8];
  const float* bfu   = (const float*)d_in[9];
  const float* Wo    = (const float*)d_in[10];
  const float* bo    = (const float*)d_in[11];
  float* out = (float*)d_out;

  const size_t rows = (size_t)LSEQ * LSEQ;
  _Float16* q  = (_Float16*)d_ws;
  _Float16* k  = q  + rows * CH;
  _Float16* v  = k  + rows * CH;
  _Float16* fu = v  + rows * CH;
  _Float16* ao = fu + rows * CH;
  float* bias  = (float*)(ao + rows * CH);

  const size_t smemA = (size_t)3 * 128 * 128 * sizeof(_Float16);                 // 96 KB
  const size_t smemB = (size_t)3 * 256 * 32 * sizeof(_Float16)
                     + (size_t)8 * 16 * 256 * sizeof(float);                     // 176 KB
  const size_t smemC = (size_t)2 * 128 * 128 * sizeof(_Float16)
                     + (size_t)128 * 128 * sizeof(float);                        // 128 KB

  (void)hipFuncSetAttribute((const void*)ln_proj_kernel,
      hipFuncAttributeMaxDynamicSharedMemorySize, (int)smemA);
  (void)hipFuncSetAttribute((const void*)attn_kernel,
      hipFuncAttributeMaxDynamicSharedMemorySize, (int)smemB);
  (void)hipFuncSetAttribute((const void*)gate_out_kernel,
      hipFuncAttributeMaxDynamicSharedMemorySize, (int)smemC);

  ln_proj_kernel<<<dim3(rows / 128), 256, smemA, stream>>>(
      pair, gamma, beta, Wq, Wk, Wv, Wfu, Wb, q, k, v, fu, bias);

  attn_kernel<<<dim3(LSEQ, NH, 2), 256, smemB, stream>>>(q, k, v, mask, bias, ao);

  gate_out_kernel<<<dim3(rows / 128), 256, smemC, stream>>>(ao, fu, bfu, Wo, bo, out);
}
